// SelfAttention_41455024341350
// MI455X (gfx1250) — matrix operand order repaired and checked by reading, not compiled, not run on hardware
//
#include <hip/hip_runtime.h>
#include <hip/hip_bf16.h>

// ---------------------------------------------------------------------------
// Types for CDNA5 WMMA (wave32)
// ---------------------------------------------------------------------------
typedef __bf16 bf16_t;
typedef __attribute__((ext_vector_type(16))) __bf16 v16bf;
typedef __attribute__((ext_vector_type(8)))  __bf16 v8bf;
typedef __attribute__((ext_vector_type(8)))  float  v8f;

#define HID   1024
#define NHEAD 16
#define HD    64
#define BATCH 8
#define SEQ   1024
#define MROWS (BATCH * SEQ)          // 8192

static constexpr size_t NX = (size_t)MROWS * HID;   // 8,388,608 elements
static constexpr size_t NW = (size_t)HID * HID;     // 1,048,576 elements

// ---------------------------------------------------------------------------
// Fragment loaders (CDNA5 WMMA 16x16x32 bf16 layouts, wave32)
// A-fragment (16x32, row-major source, row stride ld):
//   lane: m = lane&15, g = lane>>4 ; halves j=0..7 -> K = j + 8g,
//   halves j=8..15 -> K = j + 8 + 8g  (two 16B chunks at +8g and +16+8g)
// ---------------------------------------------------------------------------
__device__ __forceinline__ v16bf load_fragA(const bf16_t* tile, int ld, int lane) {
  const int m = lane & 15;
  const int g = (lane >> 4) & 1;
  const bf16_t* p = tile + (size_t)m * ld + (g << 3);
  v8bf lo = *reinterpret_cast<const v8bf*>(p);
  v8bf hi = *reinterpret_cast<const v8bf*>(p + 16);
  v16bf r;
#pragma unroll
  for (int i = 0; i < 8; ++i) { r[i] = lo[i]; r[i + 8] = hi[i]; }
  return r;
}

// B-fragment (32x16). Source stored "column-major" as Bt[n][k] with row
// stride ld. lane: n = lane&15, g = lane>>4 ; halves j=0..7 -> K = j + 8g,
// halves j=8..15 -> K = j + 8 + 8g.
__device__ __forceinline__ v16bf load_fragB(const bf16_t* tileT, int ld, int lane) {
  const int n = lane & 15;
  const int g = (lane >> 4) & 1;
  const bf16_t* p = tileT + (size_t)n * ld + (g << 3);
  v8bf lo = *reinterpret_cast<const v8bf*>(p);
  v8bf hi = *reinterpret_cast<const v8bf*>(p + 16);
  v16bf r;
#pragma unroll
  for (int i = 0; i < 8; ++i) { r[i] = lo[i]; r[i + 8] = hi[i]; }
  return r;
}

__device__ __forceinline__ v8f wmma_bf16(v16bf a, v16bf b, v8f c) {
  return __builtin_amdgcn_wmma_f32_16x16x32_bf16(
      /*neg_a=*/false, a, /*neg_b=*/false, b,
      /*c_mod=*/(short)0, c, /*reuse_a=*/false, /*reuse_b=*/false);
}

// ---------------------------------------------------------------------------
// Elementwise f32 -> bf16 convert
// ---------------------------------------------------------------------------
__global__ __launch_bounds__(256) void cvt_bf16(const float* __restrict__ src,
                                                bf16_t* __restrict__ dst, int n) {
  int i = blockIdx.x * blockDim.x + threadIdx.x;
  if (i < n) dst[i] = (bf16_t)src[i];
}

// Transpose-convert: W[k][n] (f32, [in,out]) -> Wt[n][k] (bf16, [out,in])
__global__ __launch_bounds__(256) void transpose_cvt(const float* __restrict__ src,
                                                     bf16_t* __restrict__ dst) {
  int n = blockIdx.x * 16 + threadIdx.x;  // output col of src
  int k = blockIdx.y * 16 + threadIdx.y;  // input row of src
  dst[(size_t)n * HID + k] = (bf16_t)src[(size_t)k * HID + n];
}

// ---------------------------------------------------------------------------
// GEMM: C[M=8192, N=1024] = A(bf16) @ Bt^T(bf16) + bias, WMMA bf16, f32 acc.
// One wave per 32x64 output strip (2 A-frags x 4 B-frags = 8 WMMAs/K-step).
// Ping-pong register double-buffering (k-loop unrolled by 2) so the next
// K-step's 12 b128 loads overlap the current 8 WMMAs with no register moves.
// Store modes:
//   0: Q/K per-head bf16  [b,h,s,64]
//   1: V transposed bf16  [b,h,64,s]
//   2: f32 natural        [m, n]
// ---------------------------------------------------------------------------
struct GFrags { v16bf a0, a1, b[4]; };

__device__ __forceinline__ void g_load(GFrags& f, const bf16_t* arow0,
                                       const bf16_t* arow1,
                                       const bf16_t* const brow[4],
                                       int kk, int lane) {
  f.a0 = load_fragA(arow0 + kk, HID, lane);
  f.a1 = load_fragA(arow1 + kk, HID, lane);
#pragma unroll
  for (int t = 0; t < 4; ++t) f.b[t] = load_fragB(brow[t] + kk, HID, lane);
}

__device__ __forceinline__ void g_mac(v8f (&acc)[2][4], const GFrags& f) {
#pragma unroll
  for (int t = 0; t < 4; ++t) acc[0][t] = wmma_bf16(f.a0, f.b[t], acc[0][t]);
#pragma unroll
  for (int t = 0; t < 4; ++t) acc[1][t] = wmma_bf16(f.a1, f.b[t], acc[1][t]);
}

__global__ __launch_bounds__(256) void gemm_bf16_wmma(
    const bf16_t* __restrict__ A, const bf16_t* __restrict__ Bt,
    const float* __restrict__ bias, void* __restrict__ dst, int mode) {
  const int lane = threadIdx.x & 31;
  const int wave = (int)((blockIdx.x * blockDim.x + threadIdx.x) >> 5);
  const int NT64 = HID / 64;              // 16 N-groups
  const int mTile = wave / NT64;          // 0..255, 32 rows each
  const int nG = wave % NT64;             // 0..15
  const int m0 = mTile * 32;
  const int n0 = nG * 64;

  v8f acc[2][4] = {};
  const bf16_t* arow0 = A + (size_t)m0 * HID;
  const bf16_t* arow1 = A + (size_t)(m0 + 16) * HID;
  const bf16_t* brow[4];
#pragma unroll
  for (int t = 0; t < 4; ++t) brow[t] = Bt + (size_t)(n0 + t * 16) * HID;

  GFrags f0, f1;
  g_load(f0, arow0, arow1, brow, 0, lane);

  for (int kk = 0; kk < HID; kk += 64) {
    // stage 0: prefetch kk+32 into f1 (always in range: kk+32 <= 992), use f0
    g_load(f1, arow0, arow1, brow, kk + 32, lane);
    __builtin_prefetch(arow0 + kk + 256, 0, 1);     // global_prefetch
    g_mac(acc, f0);
    // stage 1: prefetch kk+64 into f0 (guarded tail), use f1
    if (kk + 64 < HID) g_load(f0, arow0, arow1, brow, kk + 64, lane);
    g_mac(acc, f1);
  }

  const int nc = lane & 15;
  const int g = lane >> 4;
#pragma unroll
  for (int u = 0; u < 2; ++u) {
#pragma unroll
    for (int t = 0; t < 4; ++t) {
      const int ncol = n0 + t * 16 + nc;
      const float bvv = bias[ncol];
#pragma unroll
      for (int i = 0; i < 8; ++i) {
        const int m = m0 + u * 16 + i + 8 * g;
        const float val = acc[u][t][i] + bvv;
        if (mode == 2) {
          ((float*)dst)[(size_t)m * HID + ncol] = val;
        } else {
          const int b = m >> 10, s = m & (SEQ - 1);
          const int h = ncol >> 6, d = ncol & (HD - 1);
          if (mode == 0)
            ((bf16_t*)dst)[((size_t)(b * NHEAD + h) * SEQ + s) * HD + d] = (bf16_t)val;
          else
            ((bf16_t*)dst)[((size_t)(b * NHEAD + h) * HD + d) * SEQ + s] = (bf16_t)val;
        }
      }
    }
  }
}

// ---------------------------------------------------------------------------
// Flash attention: one wave per (b, h, 16-query tile). Keys streamed in
// blocks of 32 with online softmax; K fragments ping-pong double-buffered
// (j-loop unrolled by 2). scale = 1/sqrt(HIDDEN) = 1/32.
// Q,K: [b,h,s,64] bf16 ; Vt: [b,h,64,s] bf16 ; out merged: [b,s,h*64+d] bf16.
// ---------------------------------------------------------------------------
struct KFrags { v16bf k00, k01, k10, k11; };

__device__ __forceinline__ void k_load(KFrags& f, const bf16_t* kbh, int j, int lane) {
  const bf16_t* kb = kbh + (size_t)j * HD;
  f.k00 = load_fragB(kb, HD, lane);
  f.k01 = load_fragB(kb + 32, HD, lane);
  f.k10 = load_fragB(kb + 16 * HD, HD, lane);
  f.k11 = load_fragB(kb + 16 * HD + 32, HD, lane);
}

__device__ __forceinline__ void attn_step(
    int j, const KFrags& kf, const v16bf& a0, const v16bf& a1,
    const bf16_t* vbh, bf16_t* pb,
    float (&mrow)[8], float (&lrow)[8], v8f (&o)[4], int lane) {
  const int nc = lane & 15, g = lane >> 4;
  // scores: tile0 = keys j..j+15, tile1 = keys j+16..j+31 (2 WMMAs each)
  v8f s0 = {}, s1 = {};
  s0 = wmma_bf16(a0, kf.k00, s0);
  s0 = wmma_bf16(a1, kf.k01, s0);
  s1 = wmma_bf16(a0, kf.k10, s1);
  s1 = wmma_bf16(a1, kf.k11, s1);

  float p0[8], p1[8], alpha[8];
#pragma unroll
  for (int i = 0; i < 8; ++i) {
    const float x0 = s0[i] * 0.03125f;
    const float x1 = s1[i] * 0.03125f;
    float t = fmaxf(x0, x1);
#pragma unroll
    for (int off = 8; off; off >>= 1) t = fmaxf(t, __shfl_xor(t, off, 32));
    const float mn = fmaxf(mrow[i], t);
    alpha[i] = __expf(mrow[i] - mn);
    mrow[i] = mn;
    p0[i] = __expf(x0 - mn);
    p1[i] = __expf(x1 - mn);
    float r = p0[i] + p1[i];
#pragma unroll
    for (int off = 8; off; off >>= 1) r += __shfl_xor(r, off, 32);
    lrow[i] = lrow[i] * alpha[i] + r;
  }
#pragma unroll
  for (int t = 0; t < 4; ++t)
#pragma unroll
    for (int i = 0; i < 8; ++i) o[t][i] *= alpha[i];

  // Stage P (C layout) into LDS row-major [m][k], then reload as A-fragment.
#pragma unroll
  for (int i = 0; i < 8; ++i) {
    const int m = i + 8 * g;
    pb[m * 40 + nc]      = (bf16_t)p0[i];
    pb[m * 40 + 16 + nc] = (bf16_t)p1[i];
  }
  asm volatile("s_wait_dscnt 0x0" ::: "memory");
  const v16bf pa = load_fragA(pb, 40, lane);
#pragma unroll
  for (int t = 0; t < 4; ++t) {
    v16bf vb = load_fragB(vbh + (size_t)(t * 16) * SEQ + j, SEQ, lane);
    o[t] = wmma_bf16(pa, vb, o[t]);
  }
  asm volatile("s_wait_dscnt 0x0" ::: "memory");  // LDS reads done before overwrite
}

__global__ __launch_bounds__(256) void attn_wmma(
    const bf16_t* __restrict__ Q, const bf16_t* __restrict__ K,
    const bf16_t* __restrict__ Vt, bf16_t* __restrict__ Mrg) {
  __shared__ bf16_t pbuf[8][16 * 40];     // per-wave 16x32 P tile, padded ld=40
  const int lane = threadIdx.x & 31;
  const int wlocal = threadIdx.x >> 5;
  const int wave = (int)((blockIdx.x * blockDim.x + threadIdx.x) >> 5);
  const int bh = wave >> 6;               // 64 query tiles per (b,h)
  const int qT = wave & 63;
  const int b = bh >> 4, h = bh & (NHEAD - 1);

  const bf16_t* qbase = Q + ((size_t)bh * SEQ + qT * 16) * HD;
  const v16bf a0 = load_fragA(qbase, HD, lane);       // d = 0..31
  const v16bf a1 = load_fragA(qbase + 32, HD, lane);  // d = 32..63

  float mrow[8], lrow[8];
  v8f o[4] = {};
#pragma unroll
  for (int i = 0; i < 8; ++i) { mrow[i] = -1e30f; lrow[i] = 0.f; }

  bf16_t* pb = &pbuf[wlocal][0];
  const bf16_t* kbh = K + (size_t)bh * SEQ * HD;
  const bf16_t* vbh = Vt + (size_t)bh * HD * SEQ;

  KFrags kf0, kf1;
  k_load(kf0, kbh, 0, lane);

  for (int j = 0; j < SEQ; j += 64) {
    // stage 0: prefetch keys j+32 into kf1 (j+32 <= 992, always valid), use kf0
    k_load(kf1, kbh, j + 32, lane);
    attn_step(j, kf0, a0, a1, vbh, pb, mrow, lrow, o, lane);
    // stage 1: prefetch keys j+64 into kf0 (guarded tail), use kf1
    if (j + 64 < SEQ) k_load(kf0, kbh, j + 64, lane);
    attn_step(j + 32, kf1, a0, a1, vbh, pb, mrow, lrow, o, lane);
  }

  // normalize and store merged [b, s, h*64+d]
  const int nc = lane & 15, g = lane >> 4;
#pragma unroll
  for (int t = 0; t < 4; ++t) {
    const int d = t * 16 + nc;
#pragma unroll
    for (int i = 0; i < 8; ++i) {
      const int s = qT * 16 + i + 8 * g;
      const float val = o[t][i] / lrow[i];
      Mrg[((size_t)b * SEQ + s) * HID + h * HD + d] = (bf16_t)val;
    }
  }
}

// ---------------------------------------------------------------------------
extern "C" void kernel_launch(void* const* d_in, const int* in_sizes, int n_in,
                              void* d_out, int out_size, void* d_ws, size_t ws_size,
                              hipStream_t stream) {
  (void)in_sizes; (void)n_in; (void)out_size; (void)ws_size;
  const float* x  = (const float*)d_in[0];
  const float* wq = (const float*)d_in[1];
  const float* bq = (const float*)d_in[2];
  const float* wk = (const float*)d_in[3];
  const float* bk = (const float*)d_in[4];
  const float* wv = (const float*)d_in[5];
  const float* bv = (const float*)d_in[6];
  const float* wo = (const float*)d_in[7];
  const float* bo = (const float*)d_in[8];

  // Workspace layout (bf16): x, 4 transposed weights, Q, K, Vt, merged (~88 MB)
  bf16_t* xb  = (bf16_t*)d_ws;
  bf16_t* wqt = xb + NX;
  bf16_t* wkt = wqt + NW;
  bf16_t* wvt = wkt + NW;
  bf16_t* wot = wvt + NW;
  bf16_t* Qb  = wot + NW;
  bf16_t* Kb  = Qb + NX;
  bf16_t* Vtb = Kb + NX;
  bf16_t* Mrg = Vtb + NX;

  cvt_bf16<<<(int)(NX / 256), 256, 0, stream>>>(x, xb, (int)NX);
  dim3 tb(16, 16), tg(HID / 16, HID / 16);
  transpose_cvt<<<tg, tb, 0, stream>>>(wq, wqt);
  transpose_cvt<<<tg, tb, 0, stream>>>(wk, wkt);
  transpose_cvt<<<tg, tb, 0, stream>>>(wv, wvt);
  transpose_cvt<<<tg, tb, 0, stream>>>(wo, wot);

  // 4096 waves per GEMM = 512 blocks x 8 waves (wave32, 256 threads)
  gemm_bf16_wmma<<<512, 256, 0, stream>>>(xb, wqt, bq, Qb, 0);
  gemm_bf16_wmma<<<512, 256, 0, stream>>>(xb, wkt, bk, Kb, 0);
  gemm_bf16_wmma<<<512, 256, 0, stream>>>(xb, wvt, bv, Vtb, 1);

  attn_wmma<<<1024, 256, 0, stream>>>(Qb, Kb, Vtb, Mrg);

  gemm_bf16_wmma<<<512, 256, 0, stream>>>(Mrg, wot, bo, d_out, 2);
}
